// WeightOnlyQuantizedLinear_81097572483510
// MI455X (gfx1250) — matrix operand order repaired and checked by reading, not compiled, not run on hardware
//
#include <hip/hip_runtime.h>

typedef _Float16 v8h  __attribute__((ext_vector_type(8)));
typedef _Float16 v16h __attribute__((ext_vector_type(16)));
typedef float    v8f  __attribute__((ext_vector_type(8)));
typedef unsigned v4u  __attribute__((ext_vector_type(4)));
typedef unsigned v8u  __attribute__((ext_vector_type(8)));

#define M_DIM 8192
#define N_DIM 4096
#define K_DIM 4096
#define LDTE  80    // LDS row stride in f16 elems: 64 data + 2x8 pad (160B, 16B-aligned)

// ---------------------------------------------------------------------------
// Kernel 1: dequantize int8-coded weights (stored as int32) -> f16 W16[o][i]
// ---------------------------------------------------------------------------
__global__ __launch_bounds__(256)
void dequant_w16(const int* __restrict__ q, const float* __restrict__ sc,
                 const float* __restrict__ zp, _Float16* __restrict__ w16) {
    int idx = (blockIdx.x * 256 + threadIdx.x) * 8;
    int o   = idx >> 12;
    float s = sc[o];
    float z = zp[o];
    int4 a = *(const int4*)(q + idx);
    int4 b = *(const int4*)(q + idx + 4);
    v8h h;
    h[0] = (_Float16)(((float)a.x - z) * s);
    h[1] = (_Float16)(((float)a.y - z) * s);
    h[2] = (_Float16)(((float)a.z - z) * s);
    h[3] = (_Float16)(((float)a.w - z) * s);
    h[4] = (_Float16)(((float)b.x - z) * s);
    h[5] = (_Float16)(((float)b.y - z) * s);
    h[6] = (_Float16)(((float)b.z - z) * s);
    h[7] = (_Float16)(((float)b.w - z) * s);
    *(v8h*)(w16 + idx) = h;
}

// ---------------------------------------------------------------------------
// Kernel 2: convert activations fp32 -> f16 once (reused by 32 N-tiles)
// ---------------------------------------------------------------------------
__global__ __launch_bounds__(256)
void cvt_x16(const float* __restrict__ x, _Float16* __restrict__ x16) {
    int idx = (blockIdx.x * 256 + threadIdx.x) * 8;
    float4 a = *(const float4*)(x + idx);
    float4 b = *(const float4*)(x + idx + 4);
    v8h h;
    h[0]=(_Float16)a.x; h[1]=(_Float16)a.y; h[2]=(_Float16)a.z; h[3]=(_Float16)a.w;
    h[4]=(_Float16)b.x; h[5]=(_Float16)b.y; h[6]=(_Float16)b.z; h[7]=(_Float16)b.w;
    *(v8h*)(x16 + idx) = h;
}

// ---------------------------------------------------------------------------
// TDM: DMA a 128-row x 64-col f16 tile (row stride 4096 elems) into LDS.
// pad_enable inserts 16B after every 64B stored -> LDS row footprint 160B,
// i.e. two 32-k sub-windows of 64B each at byte offsets {0, 80} per row.
// ---------------------------------------------------------------------------
__device__ __forceinline__
void tdm_load_tile(unsigned lds_addr, unsigned long long gaddr, unsigned rows_total) {
    v4u g0;
    g0[0] = 1u;                                   // count=1, user descriptor
    g0[1] = lds_addr;                             // lds_addr
    g0[2] = (unsigned)gaddr;                      // global_addr[31:0]
    g0[3] = ((unsigned)(gaddr >> 32) & 0x01FFFFFFu) | 0x80000000u; // addr[56:32] | type=2
    v8u g1;
    g1[0] = (1u << 16)      // data_size = 2 bytes
          | (1u << 20)      // pad_enable
          | (3u << 22)      // pad_interval: every 16 DWORDs (64B)
          | (3u << 25);     // pad_amount: 4 DWORDs (16B)
    g1[1] = (K_DIM & 0xFFFFu) << 16;                                    // tensor_dim0 lo
    g1[2] = ((K_DIM >> 16) & 0xFFFFu) | ((rows_total & 0xFFFFu) << 16); // dim0 hi | dim1 lo
    g1[3] = ((rows_total >> 16) & 0xFFFFu) | (64u << 16);               // dim1 hi | tile_dim0=64
    g1[4] = 128u;                                                       // tile_dim1=128 rows
    g1[5] = (unsigned)K_DIM;                                            // tensor_dim0_stride
    g1[6] = 0u;
    g1[7] = 0u;
    asm volatile("tensor_load_to_lds %0, %1" :: "s"(g0), "s"(g1) : "memory");
}

// ---------------------------------------------------------------------------
// Kernel 3: y = x16 @ W16^T + bias (f16 WMMA, f32 acc), TDM double-buffered,
// K-step 64 (two 32-k WMMA sub-steps per barrier interval).
// 256 threads = 8 waves; macro-tile 128x128; wave tile 32x64.
// ---------------------------------------------------------------------------
__global__ __launch_bounds__(256)
void wmma_gemm(const _Float16* __restrict__ x16, const _Float16* __restrict__ w16,
               const float* __restrict__ bias, float* __restrict__ out) {
    __shared__ _Float16 As[2][128 * LDTE];   // 2 x 20KB
    __shared__ _Float16 Bs[2][128 * LDTE];   // 2 x 20KB

    const int tid   = threadIdx.x;
    const int lane  = tid & 31;
    const int wave  = tid >> 5;
    const int wm    = wave >> 1;         // 0..3 -> 32-row band
    const int wn    = wave & 1;          // 0..1 -> 64-col band
    const int l16   = lane & 15;
    const int lhalf = lane >> 4;

    const int m0 = blockIdx.y * 128;
    const int n0 = blockIdx.x * 128;

    const unsigned long long gA =
        (unsigned long long)(uintptr_t)(x16 + (size_t)m0 * K_DIM);
    const unsigned long long gB =
        (unsigned long long)(uintptr_t)(w16 + (size_t)n0 * K_DIM);
    const unsigned ldsA[2] = { (unsigned)(uintptr_t)&As[0][0],
                               (unsigned)(uintptr_t)&As[1][0] };
    const unsigned ldsB[2] = { (unsigned)(uintptr_t)&Bs[0][0],
                               (unsigned)(uintptr_t)&Bs[1][0] };

    v8f c[2][4];
    #pragma unroll
    for (int mi = 0; mi < 2; ++mi)
        #pragma unroll
        for (int ni = 0; ni < 4; ++ni)
            c[mi][ni] = (v8f){0.f,0.f,0.f,0.f,0.f,0.f,0.f,0.f};

    // prologue: DMA K-tile 0 into buffer 0 (one wave drives the TDM)
    if (wave == 0) {
        tdm_load_tile(ldsA[0], gA, M_DIM);
        tdm_load_tile(ldsB[0], gB, N_DIM);
    }

    int buf = 0;
    #pragma unroll 1
    for (int kk = 0; kk < K_DIM; kk += 64) {
        if (wave == 0) {
            if (kk + 64 < K_DIM) {
                const unsigned long long koff = (unsigned long long)(kk + 64) * 2u;
                tdm_load_tile(ldsA[buf ^ 1], gA + koff, M_DIM);
                tdm_load_tile(ldsB[buf ^ 1], gB + koff, N_DIM);
                __builtin_amdgcn_s_wait_tensorcnt(2);   // current pair landed
            } else {
                __builtin_amdgcn_s_wait_tensorcnt(0);
            }
        }
        __syncthreads();                 // current buffer visible to all waves

        #pragma unroll
        for (int s = 0; s < 2; ++s) {    // two 32-k sub-steps per buffer
            // ---- A fragments (per-lane layout matches ISA 16-bit A table):
            // lane<16 holds K {0..7,16..23}; lane>=16 holds K {8..15,24..31}
            v16h af[2];
            #pragma unroll
            for (int mi = 0; mi < 2; ++mi) {
                const _Float16* ap =
                    &As[buf][(wm*32 + mi*16 + l16) * LDTE + s*40 + lhalf*8];
                v8h lo = *(const v8h*)ap;
                v8h hi = *(const v8h*)(ap + 16);
                af[mi] = __builtin_shufflevector(lo, hi,
                            0,1,2,3,4,5,6,7,8,9,10,11,12,13,14,15);
            }

            // ---- B fragments: 8 LDS loads, single dscnt wait
            unsigned ba0 = (unsigned)(uintptr_t)
                (&Bs[buf][(wn*64 +  0 + l16) * LDTE + s*40 + lhalf*8]);
            unsigned ba1 = (unsigned)(uintptr_t)
                (&Bs[buf][(wn*64 + 16 + l16) * LDTE + s*40 + lhalf*8]);
            unsigned ba2 = (unsigned)(uintptr_t)
                (&Bs[buf][(wn*64 + 32 + l16) * LDTE + s*40 + lhalf*8]);
            unsigned ba3 = (unsigned)(uintptr_t)
                (&Bs[buf][(wn*64 + 48 + l16) * LDTE + s*40 + lhalf*8]);
            v8h t0,t1,t2,t3,t4,t5,t6,t7;
            asm volatile(
                "ds_load_b128 %0, %8\n\t"
                "ds_load_b128 %1, %8 offset:32\n\t"
                "ds_load_b128 %2, %9\n\t"
                "ds_load_b128 %3, %9 offset:32\n\t"
                "ds_load_b128 %4, %10\n\t"
                "ds_load_b128 %5, %10 offset:32\n\t"
                "ds_load_b128 %6, %11\n\t"
                "ds_load_b128 %7, %11 offset:32\n\t"
                "s_wait_dscnt 0x0"
                : "=&v"(t0), "=&v"(t1), "=&v"(t2), "=&v"(t3),
                  "=&v"(t4), "=&v"(t5), "=&v"(t6), "=&v"(t7)
                : "v"(ba0), "v"(ba1), "v"(ba2), "v"(ba3)
                : "memory");
            v16h bf[4];
            bf[0] = __builtin_shufflevector(t0, t1, 0,1,2,3,4,5,6,7,8,9,10,11,12,13,14,15);
            bf[1] = __builtin_shufflevector(t2, t3, 0,1,2,3,4,5,6,7,8,9,10,11,12,13,14,15);
            bf[2] = __builtin_shufflevector(t4, t5, 0,1,2,3,4,5,6,7,8,9,10,11,12,13,14,15);
            bf[3] = __builtin_shufflevector(t6, t7, 0,1,2,3,4,5,6,7,8,9,10,11,12,13,14,15);

            // ---- 8 WMMAs per sub-step (32x64 wave tile, f32 accumulate)
            #pragma unroll
            for (int mi = 0; mi < 2; ++mi)
                #pragma unroll
                for (int ni = 0; ni < 4; ++ni)
                    c[mi][ni] = __builtin_amdgcn_wmma_f32_16x16x32_f16(
                        false, af[mi], false, bf[ni],
                        (short)0, c[mi][ni], false, false);
        }

        __syncthreads();                 // buffer free before TDM refills it
        buf ^= 1;
    }

    // ---- epilogue: C layout (VGPR r, lane): M = r + 8*lhalf, N = l16
    #pragma unroll
    for (int mi = 0; mi < 2; ++mi) {
        #pragma unroll
        for (int ni = 0; ni < 4; ++ni) {
            const int gN = n0 + wn*64 + ni*16 + l16;
            const float bv = bias[gN];
            float* op = out + (size_t)(m0 + wm*32 + mi*16 + lhalf*8) * N_DIM + gN;
            #pragma unroll
            for (int r = 0; r < 8; ++r)
                op[(size_t)r * N_DIM] = c[mi][ni][r] + bv;
        }
    }
}

// ---------------------------------------------------------------------------
extern "C" void kernel_launch(void* const* d_in, const int* in_sizes, int n_in,
                              void* d_out, int out_size, void* d_ws, size_t ws_size,
                              hipStream_t stream) {
    const float* x    = (const float*)d_in[0];
    const int*   qw   = (const int*)  d_in[1];
    const float* sc   = (const float*)d_in[2];
    const float* zp   = (const float*)d_in[3];
    const float* bias = (const float*)d_in[4];
    float*       out  = (float*)d_out;

    _Float16* w16 = (_Float16*)d_ws;                       // 32 MB
    _Float16* x16 = w16 + (size_t)N_DIM * K_DIM;           // +64 MB

    dequant_w16<<<(N_DIM * K_DIM) / (256 * 8), 256, 0, stream>>>(qw, sc, zp, w16);
    cvt_x16   <<<(M_DIM * K_DIM) / (256 * 8), 256, 0, stream>>>(x, x16);

    dim3 grid(N_DIM / 128, M_DIM / 128);
    wmma_gemm<<<grid, 256, 0, stream>>>(x16, w16, bias, out);
}
